// Discriminator_40355512713723
// MI455X (gfx1250) — matrix operand order repaired and checked by reading, not compiled, not run on hardware
//
#include <hip/hip_runtime.h>

#define B_    128
#define T_    256
#define HD_   512
#define BT_   (B_*T_)
#define NREAL 32
#define NCAT  4
#define NCLS  100
#define EMB_  8
#define X0W   (NREAL + NCAT)   /* 36 */
#define CMBW  64
#define G4    (4*HD_)          /* 2048 */

typedef unsigned short u16;
typedef __attribute__((ext_vector_type(16))) __bf16 bf16x16;
typedef __attribute__((ext_vector_type(8)))  float  floatx8;

union Frag { bf16x16 v; uint4 q[2]; };

__device__ __forceinline__ u16 f2bf(float f) {
  unsigned u = __float_as_uint(f);
  u += 0x7FFFu + ((u >> 16) & 1u);
  return (u16)(u >> 16);
}
__device__ __forceinline__ float bf2f(u16 h) { return __uint_as_float(((unsigned)h) << 16); }
__device__ __forceinline__ float sigm(float x) { return 1.0f / (1.0f + __expf(-x)); }

// ---- CDNA5 async global->LDS copies (ASYNCcnt) -------------------------------
// INST_OFFSET applies to BOTH the LDS and global address (ISA 08 §4.4), so a
// 32B copy is two b128 issues sharing one address pair.
__device__ __forceinline__ void async_cp32(unsigned ldsOff, const void* g) {
  asm volatile("global_load_async_to_lds_b128 %0, %1, off\n\t"
               "global_load_async_to_lds_b128 %0, %1, off offset:16"
               :: "v"(ldsOff), "v"(g) : "memory");
}
template <int N>
__device__ __forceinline__ void wait_async() {
  asm volatile("s_wait_asynccnt %0" :: "i"(N) : "memory");
}
// Generic LDS pointers carry the LDS byte offset in addr[31:0].
__device__ __forceinline__ unsigned lds_off(const void* p) {
  return (unsigned)(size_t)p;
}

// ---------------- small utility kernels ----------------
__global__ void f2bf_transpose_kernel(const float* __restrict__ s, u16* __restrict__ d,
                                      int K, int N) {
  // d[n*K + k] = bf16(s[k*N + n])   (one-time weight prep)
  int i = blockIdx.x * blockDim.x + threadIdx.x;
  if (i >= K * N) return;
  int n = i / K, k = i - n * K;
  d[i] = f2bf(s[(size_t)k * N + n]);
}

__global__ void bias_sum_kernel(const float* __restrict__ a, const float* __restrict__ b,
                                float* __restrict__ d, int n) {
  int i = blockIdx.x * blockDim.x + threadIdx.x;
  if (i < n) d[i] = a[i] + b[i];
}

__global__ void embed_kernel(const float* __restrict__ x0, const float* __restrict__ emb,
                             u16* __restrict__ cmb) {
  int i = blockIdx.x * blockDim.x + threadIdx.x;   // over BT_*CMBW
  if (i >= BT_ * CMBW) return;
  int bt = i >> 6, j = i & 63;
  float v;
  if (j < NREAL) {
    v = x0[bt * X0W + j];
  } else {
    int c = (j - NREAL) >> 3, e = (j - NREAL) & 7;
    int idx = (int)x0[bt * X0W + NREAL + c];
    v = emb[((size_t)c * NCLS + idx) * EMB_ + e];
  }
  cmb[i] = f2bf(v);
}

__global__ void init_out_kernel(float* __restrict__ out, const float* __restrict__ b3) {
  int i = blockIdx.x * blockDim.x + threadIdx.x;
  if (i < BT_) out[i] = b3[0];
}

__global__ void zero_state_kernel(u16* __restrict__ h, float* __restrict__ c) {
  int i = blockIdx.x * blockDim.x + threadIdx.x;
  if (i < B_ * HD_) { h[i] = 0; c[i] = 0.0f; }
}

// ---------------- bf16 WMMA GEMM: out = act(A[MxK] @ Wt[NxK]^T + bias) -------
// Wt pre-transposed. 256 thr = 8 waves (4 M x 2 N), wave tile 32x64, block tile
// 128x128, K step 32, async double-buffered LDS. 8 wmma per wave per K-step.
template <int ACT>
__global__ __launch_bounds__(256) void gemm_bf16_kernel(
    const u16* __restrict__ A, const u16* __restrict__ Wt,
    const float* __restrict__ bias, u16* __restrict__ out,
    int M, int N, int K) {
  __shared__ u16 As[2][128][40];
  __shared__ u16 Bs[2][128][40];
  const int tid = threadIdx.x;
  const int wave = tid >> 5, lane = tid & 31;
  const int lrow = lane & 15, half = lane >> 4;
  const int waveM = wave & 3, waveN = wave >> 2;
  const int mBase = blockIdx.x * 128, nBase = blockIdx.y * 128;

  const int srow = tid >> 1, scg = (tid & 1) * 16;   // both tiles: 128x32, 32B/thread
  const u16* aSrc = A + (size_t)(mBase + srow) * K + scg;
  const u16* bSrc = Wt + (size_t)(nBase + srow) * K + scg;

  floatx8 acc[4][2];
  for (int f = 0; f < 4; ++f)
    for (int ms = 0; ms < 2; ++ms)
      for (int r = 0; r < 8; ++r) acc[f][ms][r] = 0.0f;

  const int nIter = K >> 5;
  async_cp32(lds_off(&As[0][srow][scg]), aSrc);
  async_cp32(lds_off(&Bs[0][srow][scg]), bSrc);

  for (int it = 0; it < nIter; ++it) {
    const int buf = it & 1;
    if (it + 1 < nIter) {  // prefetch next tile (4 issues in flight)
      const int k0 = (it + 1) << 5;
      async_cp32(lds_off(&As[buf ^ 1][srow][scg]), aSrc + k0);
      async_cp32(lds_off(&Bs[buf ^ 1][srow][scg]), bSrc + k0);
      wait_async<4>();
    } else {
      wait_async<0>();
    }
    __syncthreads();

    Frag af[2];
    for (int ms = 0; ms < 2; ++ms) {
      int m = waveM * 32 + ms * 16 + lrow;
      af[ms].q[0] = *(const uint4*)&As[buf][m][half * 8];
      af[ms].q[1] = *(const uint4*)&As[buf][m][16 + half * 8];
    }
    for (int f = 0; f < 4; ++f) {
      int nl = waveN * 64 + f * 16 + lrow;
      Frag bf_;
      bf_.q[0] = *(const uint4*)&Bs[buf][nl][half * 8];
      bf_.q[1] = *(const uint4*)&Bs[buf][nl][16 + half * 8];
      for (int ms = 0; ms < 2; ++ms)
        acc[f][ms] = __builtin_amdgcn_wmma_f32_16x16x32_bf16(
            false, af[ms].v, false, bf_.v, (short)0, acc[f][ms], false, false);
    }
    __syncthreads();
  }
  for (int f = 0; f < 4; ++f) {
    int n = nBase + waveN * 64 + f * 16 + lrow;
    float bv = bias[n];
    for (int ms = 0; ms < 2; ++ms)
      for (int r = 0; r < 8; ++r) {
        int m = mBase + waveM * 32 + ms * 16 + half * 8 + r;
        float v = acc[f][ms][r] + bv;
        if (ACT) v = (v >= 0.0f) ? v : 0.1f * v;
        out[(size_t)m * N + n] = f2bf(v);
      }
  }
}

// ---------------- fused LSTM step ----------------
// 16 blocks; block j owns hidden columns [32j, 32j+32) for all 4 gates, all 128
// batch rows. Wave tile 32(b) x 64(n): the 64-wide N tile is 4 gate fragments of
// the SAME 16 hidden columns (LDS row nl = cHalf*64 + g*16 + lr holds weight
// column g*512 + jBase + cHalf*16 + lr), so each lane ends up with all 4 gate
// pre-activations for its (b, nh) element. G = xg_t + h_in @ Wh; gates -> c,h;
// fused W3 projection. Wht pre-transposed: Wht[col][k].
__global__ __launch_bounds__(256) void lstm_step_kernel(
    const u16* __restrict__ xg, int tsrc,
    const u16* __restrict__ Wht,
    const u16* __restrict__ h_in, u16* __restrict__ h_out,
    float* __restrict__ cst,
    const float* __restrict__ W3p,
    float* __restrict__ out, int tout) {
  __shared__ u16 As[2][128][40];
  __shared__ u16 Bs[2][128][40];
  const int tid = threadIdx.x;
  const int wave = tid >> 5, lane = tid & 31;
  const int lrow = lane & 15, half = lane >> 4;
  const int waveM = wave & 3, waveN = wave >> 2;
  const int jBase = blockIdx.x * 32;

  const int srow = tid >> 1, scg = (tid & 1) * 16;   // both tiles: 128x32, 32B/thread
  // LDS B row -> weight column mapping (gate-major inside each 64-wide half)
  const int bCol = ((srow >> 4) & 3) * HD_ + jBase + (srow >> 6) * 16 + (srow & 15);
  const u16* aSrc = h_in + (size_t)srow * HD_ + scg;
  const u16* bSrc = Wht + (size_t)bCol * HD_ + scg;

  floatx8 acc[4][2];
  for (int g = 0; g < 4; ++g)
    for (int ms = 0; ms < 2; ++ms)
      for (int r = 0; r < 8; ++r) {
        int b = waveM * 32 + ms * 16 + half * 8 + r;
        int col = g * HD_ + jBase + waveN * 16 + lrow;
        acc[g][ms][r] = bf2f(xg[(size_t)(b * T_ + tsrc) * G4 + col]);
      }

  const int nIter = HD_ >> 5;
  async_cp32(lds_off(&As[0][srow][scg]), aSrc);
  async_cp32(lds_off(&Bs[0][srow][scg]), bSrc);

  for (int it = 0; it < nIter; ++it) {
    const int buf = it & 1;
    if (it + 1 < nIter) {  // prefetch next tile (4 issues in flight)
      const int k0 = (it + 1) << 5;
      async_cp32(lds_off(&As[buf ^ 1][srow][scg]), aSrc + k0);
      async_cp32(lds_off(&Bs[buf ^ 1][srow][scg]), bSrc + k0);
      wait_async<4>();
    } else {
      wait_async<0>();
    }
    __syncthreads();

    Frag af[2];
    for (int ms = 0; ms < 2; ++ms) {
      int m = waveM * 32 + ms * 16 + lrow;
      af[ms].q[0] = *(const uint4*)&As[buf][m][half * 8];
      af[ms].q[1] = *(const uint4*)&As[buf][m][16 + half * 8];
    }
    for (int g = 0; g < 4; ++g) {
      int nl = waveN * 64 + g * 16 + lrow;
      Frag bf_;
      bf_.q[0] = *(const uint4*)&Bs[buf][nl][half * 8];
      bf_.q[1] = *(const uint4*)&Bs[buf][nl][16 + half * 8];
      for (int ms = 0; ms < 2; ++ms)
        acc[g][ms] = __builtin_amdgcn_wmma_f32_16x16x32_bf16(
            false, af[ms].v, false, bf_.v, (short)0, acc[g][ms], false, false);
    }
    __syncthreads();
  }

  // gate math + state update + fused projection
  const int nh = jBase + waveN * 16 + lrow;
  const float w3v = W3p[nh];
  for (int ms = 0; ms < 2; ++ms)
    for (int r = 0; r < 8; ++r) {
      int b = waveM * 32 + ms * 16 + half * 8 + r;
      float fg = sigm(acc[0][ms][r]);
      float ig = sigm(acc[1][ms][r]);
      float ag = tanhf(acc[2][ms][r]);
      float og = sigm(acc[3][ms][r]);
      size_t ci = (size_t)b * HD_ + nh;
      float cv = fg * cst[ci] + ig * ag;
      cst[ci] = cv;
      float hv = og * tanhf(cv);
      h_out[ci] = f2bf(hv);
      float p = hv * w3v;
      for (int s = 1; s < 16; s <<= 1) p += __shfl_xor(p, s);
      if (lrow == 0) atomicAdd(&out[b * T_ + tout], p);
    }
}

// ---------------- host ----------------
extern "C" void kernel_launch(void* const* d_in, const int* in_sizes, int n_in,
                              void* d_out, int out_size, void* d_ws, size_t ws_size,
                              hipStream_t stream) {
  const float* x0   = (const float*)d_in[0];
  const float* emb  = (const float*)d_in[1];
  const float* W1   = (const float*)d_in[2];
  const float* b1   = (const float*)d_in[3];
  const float* W2   = (const float*)d_in[4];
  const float* b2   = (const float*)d_in[5];
  const float* Wi_f = (const float*)d_in[6];
  const float* bi_f = (const float*)d_in[7];
  const float* Wh_f = (const float*)d_in[8];
  const float* bh_f = (const float*)d_in[9];
  const float* Wi_r = (const float*)d_in[10];
  const float* bi_r = (const float*)d_in[11];
  const float* Wh_r = (const float*)d_in[12];
  const float* bh_r = (const float*)d_in[13];
  const float* W3   = (const float*)d_in[14];
  const float* b3   = (const float*)d_in[15];
  float* out = (float*)d_out;

  char* ws = (char*)d_ws;
  size_t off = 0;
  auto take = [&](size_t bytes) -> void* {
    void* p = ws + off;
    off = (off + bytes + 255) & ~(size_t)255;
    return p;
  };
  u16*   w1t   = (u16*)take((size_t)CMBW * HD_ * 2);   // [HD_][CMBW]
  u16*   w2t   = (u16*)take((size_t)HD_ * HD_ * 2);    // [HD_][HD_]
  u16*   witf  = (u16*)take((size_t)HD_ * G4 * 2);     // [G4][HD_]
  u16*   whtf  = (u16*)take((size_t)HD_ * G4 * 2);
  u16*   witr  = (u16*)take((size_t)HD_ * G4 * 2);
  u16*   whtr  = (u16*)take((size_t)HD_ * G4 * 2);
  float* biasf = (float*)take((size_t)G4 * 4);
  float* biasr = (float*)take((size_t)G4 * 4);
  u16*   cmb   = (u16*)take((size_t)BT_ * CMBW * 2);
  u16*   x1    = (u16*)take((size_t)BT_ * HD_ * 2);
  u16*   x2    = (u16*)take((size_t)BT_ * HD_ * 2);
  u16*   xgb   = (u16*)take((size_t)BT_ * G4 * 2);
  u16*   h0    = (u16*)take((size_t)B_ * HD_ * 2);
  u16*   h1    = (u16*)take((size_t)B_ * HD_ * 2);
  float* cst   = (float*)take((size_t)B_ * HD_ * 4);
  (void)ws_size; (void)in_sizes; (void)n_in; (void)out_size;

  const int THR = 256;
  auto blks = [](size_t n) { return (unsigned)((n + 255) / 256); };

  // weight prep: convert to bf16 AND transpose (so GEMM B-tiles are contiguous)
  f2bf_transpose_kernel<<<blks((size_t)CMBW * HD_), THR, 0, stream>>>(W1, w1t, CMBW, HD_);
  f2bf_transpose_kernel<<<blks((size_t)HD_ * HD_), THR, 0, stream>>>(W2, w2t, HD_, HD_);
  f2bf_transpose_kernel<<<blks((size_t)HD_ * G4), THR, 0, stream>>>(Wi_f, witf, HD_, G4);
  f2bf_transpose_kernel<<<blks((size_t)HD_ * G4), THR, 0, stream>>>(Wh_f, whtf, HD_, G4);
  f2bf_transpose_kernel<<<blks((size_t)HD_ * G4), THR, 0, stream>>>(Wi_r, witr, HD_, G4);
  f2bf_transpose_kernel<<<blks((size_t)HD_ * G4), THR, 0, stream>>>(Wh_r, whtr, HD_, G4);
  bias_sum_kernel<<<blks(G4), THR, 0, stream>>>(bi_f, bh_f, biasf, G4);
  bias_sum_kernel<<<blks(G4), THR, 0, stream>>>(bi_r, bh_r, biasr, G4);

  // frontend
  embed_kernel<<<blks((size_t)BT_ * CMBW), THR, 0, stream>>>(x0, emb, cmb);
  gemm_bf16_kernel<1><<<dim3(BT_ / 128, HD_ / 128), THR, 0, stream>>>(
      cmb, w1t, b1, x1, BT_, HD_, CMBW);
  gemm_bf16_kernel<1><<<dim3(BT_ / 128, HD_ / 128), THR, 0, stream>>>(
      x1, w2t, b2, x2, BT_, HD_, HD_);

  init_out_kernel<<<blks(BT_), THR, 0, stream>>>(out, b3);

  // forward LSTM: xg precompute then 256 fused steps
  gemm_bf16_kernel<0><<<dim3(BT_ / 128, G4 / 128), THR, 0, stream>>>(
      x2, witf, biasf, xgb, BT_, G4, HD_);
  zero_state_kernel<<<blks((size_t)B_ * HD_), THR, 0, stream>>>(h0, cst);
  for (int t = 0; t < T_; ++t) {
    u16* hin = (t & 1) ? h1 : h0;
    u16* hout = (t & 1) ? h0 : h1;
    lstm_step_kernel<<<HD_ / 32, THR, 0, stream>>>(
        xgb, t, whtf, hin, hout, cst, W3, out, t);
  }

  // reverse LSTM: same xg buffer reused; step t consumes x2[b, T-1-t], writes out[b, t]
  gemm_bf16_kernel<0><<<dim3(BT_ / 128, G4 / 128), THR, 0, stream>>>(
      x2, witr, biasr, xgb, BT_, G4, HD_);
  zero_state_kernel<<<blks((size_t)B_ * HD_), THR, 0, stream>>>(h0, cst);
  for (int t = 0; t < T_; ++t) {
    u16* hin = (t & 1) ? h1 : h0;
    u16* hout = (t & 1) ? h0 : h1;
    lstm_step_kernel<<<HD_ / 32, THR, 0, stream>>>(
        xgb, T_ - 1 - t, whtr, hin, hout, cst, W3 + HD_, out, t);
  }
}
